// MultiGraphAttention_7121055777173
// MI455X (gfx1250) — compile-verified, matrix operand order checked by reading, not run on hardware
//
#include <hip/hip_runtime.h>
#include <math.h>

// ---------------------------------------------------------------------------
// MultiGraphAttention for MI455X (gfx1250, wave32, WMMA)
//
// Roofline: adjs [4,4096,4096] i32 = 256MB dominates (~11us @ 23.3TB/s).
// Both GEMMs run on v_wmma_f32_16x16x32_bf16 with f32 accumulation.
// Attention matrix is built fused with the adjs stream (one row per block,
// row softmax in LDS) and stored bf16 (32MB) to feed the final WMMA GEMM.
// Final GEMM: one wave computes a 16x128 stripe -> A loaded once per k-step,
// 8 WMMAs issued against the L2-resident B panel.
// ---------------------------------------------------------------------------

typedef __attribute__((ext_vector_type(16))) __bf16          v16bf;
typedef __attribute__((ext_vector_type(16))) unsigned short  v16us;
typedef __attribute__((ext_vector_type(8)))  float           v8f;
typedef __attribute__((ext_vector_type(4)))  unsigned int    v4u;

union Bf16x16 {
    v16us us;
    v16bf bf;
    v4u   q[2];   // q[0] = elements 0..7, q[1] = elements 8..15
};

#define N_NODES 4096
#define IN_DIM  512
#define OUT_DIM 128
#define K_NETS  4
#define ALPHA   0.2f

__device__ __forceinline__ unsigned short f2bf(float f) {
    unsigned int u = __float_as_uint(f);
    unsigned int r = u + 0x7FFFu + ((u >> 16) & 1u);   // round-to-nearest-even
    return (unsigned short)(r >> 16);
}

// ---------------------------------------------------------------------------
// Kernel A: Wh = h @ W   [4096,512]x[512,128], bf16 WMMA, f32 accum.
// A-operand (h) is K-contiguous -> float4 loads. Emits Wh (f32, row-major)
// and WhbT (bf16, transposed [128,4096]) so kernel D's B loads are
// K-contiguous b128.
// ---------------------------------------------------------------------------
__global__ __launch_bounds__(256) void gemm_wh_kernel(
    const float* __restrict__ h, const float* __restrict__ W,
    float* __restrict__ Wh, unsigned short* __restrict__ WhbT)
{
    const int lane   = threadIdx.x & 31;
    const int wid    = threadIdx.x >> 5;
    const int tile   = blockIdx.x * 8 + wid;   // 2048 tiles total
    const int tm     = tile >> 3;              // 256 M-tiles
    const int tn     = tile & 7;               // 8  N-tiles
    const int laneHi = lane >> 4;
    const int lane16 = lane & 15;
    const int arow   = tm * 16 + lane16;       // A row for this lane
    const int bcol   = tn * 16 + lane16;       // B col for this lane
    const float* hrow = h + (size_t)arow * IN_DIM;

    v8f c = {};
    for (int kb = 0; kb < IN_DIM; kb += 32) {
        Bf16x16 A, B;
        // A: two contiguous 8-float runs: K = kb + laneHi*8 + {0..7} and +16
#pragma unroll
        for (int g = 0; g < 2; ++g) {
            const int Kb = kb + g * 16 + laneHi * 8;
            float4 v0 = *reinterpret_cast<const float4*>(hrow + Kb);
            float4 v1 = *reinterpret_cast<const float4*>(hrow + Kb + 4);
            A.us[g * 8 + 0] = f2bf(v0.x); A.us[g * 8 + 1] = f2bf(v0.y);
            A.us[g * 8 + 2] = f2bf(v0.z); A.us[g * 8 + 3] = f2bf(v0.w);
            A.us[g * 8 + 4] = f2bf(v1.x); A.us[g * 8 + 5] = f2bf(v1.y);
            A.us[g * 8 + 6] = f2bf(v1.z); A.us[g * 8 + 7] = f2bf(v1.w);
        }
        // B: W is row-major [K, 128]; K strided -> per-element loads (tiny kernel)
#pragma unroll
        for (int j = 0; j < 16; ++j) {
            int K = kb + ((j >> 3) << 4) + (j & 7) + laneHi * 8;
            B.us[j] = f2bf(W[K * OUT_DIM + bcol]);
        }
        c = __builtin_amdgcn_wmma_f32_16x16x32_bf16(
                false, A.bf, false, B.bf, (short)0, c, false, false);
    }
#pragma unroll
    for (int r = 0; r < 8; ++r) {
        int row = tm * 16 + r + laneHi * 8;
        int col = tn * 16 + lane16;
        float v = c[r];
        Wh[row * OUT_DIM + col]   = v;
        WhbT[col * N_NODES + row] = f2bf(v);
    }
}

// ---------------------------------------------------------------------------
// Kernel B: Wh1[k,n] = Wh[n,:]·a[k,0,:],  Wh2[k,n] = Wh[n,:]·a[k,1,:]
// Tiny: one thread per node, 8 dot products of length 128 (float4 loads).
// ---------------------------------------------------------------------------
__global__ __launch_bounds__(256) void proj_kernel(
    const float* __restrict__ Wh, const float* __restrict__ a,
    float* __restrict__ Wh1, float* __restrict__ Wh2)
{
    int n = blockIdx.x * blockDim.x + threadIdx.x;
    if (n >= N_NODES) return;
    float acc[2 * K_NETS] = {};
    const float4* whr = reinterpret_cast<const float4*>(Wh + (size_t)n * OUT_DIM);
#pragma unroll 4
    for (int d4 = 0; d4 < OUT_DIM / 4; ++d4) {
        float4 v = whr[d4];
#pragma unroll
        for (int t = 0; t < 2 * K_NETS; ++t) {
            float4 av = reinterpret_cast<const float4*>(a + t * OUT_DIM)[d4];
            acc[t] += v.x * av.x + v.y * av.y + v.z * av.z + v.w * av.w;
        }
    }
#pragma unroll
    for (int k = 0; k < K_NETS; ++k) {
        Wh1[k * N_NODES + n] = acc[2 * k + 0];
        Wh2[k * N_NODES + n] = acc[2 * k + 1];
    }
}

// ---------------------------------------------------------------------------
// Block reductions (256 threads = 8 wave32)
// ---------------------------------------------------------------------------
__device__ __forceinline__ float blockReduceMax(float v, float* red) {
#pragma unroll
    for (int o = 16; o > 0; o >>= 1) v = fmaxf(v, __shfl_down(v, o, 32));
    int w = threadIdx.x >> 5, l = threadIdx.x & 31;
    __syncthreads();
    if (l == 0) red[w] = v;
    __syncthreads();
    v = (l < 8) ? red[l] : -3.4e38f;
#pragma unroll
    for (int o = 4; o > 0; o >>= 1) v = fmaxf(v, __shfl_down(v, o, 32));
    return __shfl(v, 0, 32);
}

__device__ __forceinline__ float blockReduceSum(float v, float* red) {
#pragma unroll
    for (int o = 16; o > 0; o >>= 1) v += __shfl_down(v, o, 32);
    int w = threadIdx.x >> 5, l = threadIdx.x & 31;
    __syncthreads();
    if (l == 0) red[w] = v;
    __syncthreads();
    v = (l < 8) ? red[l] : 0.f;
#pragma unroll
    for (int o = 4; o > 0; o >>= 1) v += __shfl_down(v, o, 32);
    return __shfl(v, 0, 32);
}

// ---------------------------------------------------------------------------
// Kernel C: one block per row n. Streams adjs[k,n,:] (int4, 128-bit
// coalesced, with global_prefetch one chunk ahead), builds the weighted
// masked leaky-relu logits in LDS, row-softmax, writes attention row bf16.
// This kernel carries the 256MB adjs traffic -> pure streaming.
// ---------------------------------------------------------------------------
__global__ __launch_bounds__(256) void attn_rows_kernel(
    const int* __restrict__ adjs, const float* __restrict__ Wh1,
    const float* __restrict__ Wh2, const float* __restrict__ nw,
    unsigned short* __restrict__ attb)
{
    __shared__ float row[N_NODES];   // 16 KB of the 320 KB WGP LDS
    __shared__ float red[8];

    const int n   = blockIdx.x;
    const int tid = threadIdx.x;

    // softmax over the 4 network weights (redundant per-thread, trivially cheap)
    float w0 = nw[0], w1 = nw[1], w2 = nw[2], w3 = nw[3];
    float wm = fmaxf(fmaxf(w0, w1), fmaxf(w2, w3));
    float e0 = __expf(w0 - wm), e1 = __expf(w1 - wm);
    float e2 = __expf(w2 - wm), e3 = __expf(w3 - wm);
    float es = e0 + e1 + e2 + e3;
    float wts[K_NETS] = { e0 / es, e1 / es, e2 / es, e3 / es };

    float a1[K_NETS];
#pragma unroll
    for (int k = 0; k < K_NETS; ++k) a1[k] = Wh1[k * N_NODES + n];

    float lmax = -3.4e38f;
    for (int m0 = tid * 4; m0 < N_NODES; m0 += 256 * 4) {
        float acc[4] = { 0.f, 0.f, 0.f, 0.f };
#pragma unroll
        for (int k = 0; k < K_NETS; ++k) {
            const int* ap = adjs + (((size_t)k) << 24) + (((size_t)n) << 12) + m0;
            if (m0 + 1024 < N_NODES)
                __builtin_prefetch(ap + 1024, 0, 0);   // global_prefetch_b8
            int4 adj = *reinterpret_cast<const int4*>(ap);
            float4 w2v = *reinterpret_cast<const float4*>(Wh2 + (k << 12) + m0);
            float x0 = a1[k] + w2v.x; x0 = x0 >= 0.f ? x0 : ALPHA * x0;
            float x1 = a1[k] + w2v.y; x1 = x1 >= 0.f ? x1 : ALPHA * x1;
            float x2 = a1[k] + w2v.z; x2 = x2 >= 0.f ? x2 : ALPHA * x2;
            float x3 = a1[k] + w2v.w; x3 = x3 >= 0.f ? x3 : ALPHA * x3;
            if (adj.x > 0) acc[0] += wts[k] * x0;
            if (adj.y > 0) acc[1] += wts[k] * x1;
            if (adj.z > 0) acc[2] += wts[k] * x2;
            if (adj.w > 0) acc[3] += wts[k] * x3;
        }
#pragma unroll
        for (int j = 0; j < 4; ++j) {
            row[m0 + j] = acc[j];
            lmax = fmaxf(lmax, acc[j]);
        }
    }

    float rmax = blockReduceMax(lmax, red);

    float lsum = 0.f;
    for (int m0 = tid * 4; m0 < N_NODES; m0 += 256 * 4) {
#pragma unroll
        for (int j = 0; j < 4; ++j) {
            float v = __expf(row[m0 + j] - rmax);
            row[m0 + j] = v;
            lsum += v;
        }
    }

    float inv = 1.0f / blockReduceSum(lsum, red);

    const size_t ob = ((size_t)n) << 12;
    for (int m0 = tid * 4; m0 < N_NODES; m0 += 256 * 4) {
#pragma unroll
        for (int j = 0; j < 4; ++j)
            attb[ob + m0 + j] = f2bf(row[m0 + j] * inv);
    }
}

// ---------------------------------------------------------------------------
// Kernel D: out = elu(att_bf16 @ Wh_bf16)  [4096,4096]x[4096,128]
// One wave computes a 16x128 stripe: A loaded once per k-step (2 b128),
// then 8 WMMAs against the 8 B column-tiles (whbT is 1MB -> L2/L0 resident).
// 8 v8f accumulators = 64 VGPRs; ~90 VGPRs total.
// ---------------------------------------------------------------------------
__global__ __launch_bounds__(256) void gemm_out_kernel(
    const unsigned short* __restrict__ attb,
    const unsigned short* __restrict__ whbT,
    float* __restrict__ out)
{
    const int lane   = threadIdx.x & 31;
    const int wid    = threadIdx.x >> 5;
    const int tm     = blockIdx.x * 8 + wid;   // 256 M-tiles, 32 blocks
    const int laneHi = lane >> 4;
    const int lane16 = lane & 15;

    // lane base pointers; 16-byte aligned (row stride 4096, kb mult of 32)
    const unsigned short* ap = attb + (((size_t)(tm * 16 + lane16)) << 12) + laneHi * 8;
    const unsigned short* bp = whbT + (((size_t)lane16) << 12) + laneHi * 8;

    v8f c[8];
#pragma unroll
    for (int t = 0; t < 8; ++t) c[t] = (v8f){};

    for (int kb = 0; kb < N_NODES; kb += 32) {
        Bf16x16 A;
        A.q[0] = *reinterpret_cast<const v4u*>(ap + kb);        // K = kb+off+{0..7}
        A.q[1] = *reinterpret_cast<const v4u*>(ap + kb + 16);   // K = kb+16+off+{0..7}
#pragma unroll
        for (int tn = 0; tn < 8; ++tn) {
            const unsigned short* bt = bp + (((size_t)(tn * 16)) << 12);
            Bf16x16 B;
            B.q[0] = *reinterpret_cast<const v4u*>(bt + kb);
            B.q[1] = *reinterpret_cast<const v4u*>(bt + kb + 16);
            c[tn] = __builtin_amdgcn_wmma_f32_16x16x32_bf16(
                        false, A.bf, false, B.bf, (short)0, c[tn], false, false);
        }
    }
#pragma unroll
    for (int tn = 0; tn < 8; ++tn) {
#pragma unroll
        for (int r = 0; r < 8; ++r) {
            int row = tm * 16 + r + laneHi * 8;
            int col = tn * 16 + lane16;
            float x = c[tn][r];
            out[row * OUT_DIM + col] = x > 0.f ? x : (__expf(x) - 1.0f);  // ELU
        }
    }
}

// ---------------------------------------------------------------------------
// Launch
// ---------------------------------------------------------------------------
extern "C" void kernel_launch(void* const* d_in, const int* in_sizes, int n_in,
                              void* d_out, int out_size, void* d_ws, size_t ws_size,
                              hipStream_t stream) {
    const float* h    = (const float*)d_in[0];   // [4096,512]
    const int*   adjs = (const int*)  d_in[1];   // [4,4096,4096]
    const float* W    = (const float*)d_in[2];   // [512,128]
    const float* a    = (const float*)d_in[3];   // [1024,1]
    const float* nw   = (const float*)d_in[4];   // [4,1]
    float*       out  = (float*)d_out;           // [4096,128]

    char* ws = (char*)d_ws;
    float*          Wh   = (float*)(ws + 0);                    // 2 MB
    unsigned short* WhbT = (unsigned short*)(ws + 2097152);     // 1 MB
    float*          Wh1  = (float*)(ws + 3145728);              // 64 KB
    float*          Wh2  = (float*)(ws + 3211264);              // 64 KB
    unsigned short* attb = (unsigned short*)(ws + 4194304);     // 32 MB

    gemm_wh_kernel  <<<256, 256, 0, stream>>>(h, W, Wh, WhbT);
    proj_kernel     <<<16,  256, 0, stream>>>(Wh, a, Wh1, Wh2);
    attn_rows_kernel<<<N_NODES, 256, 0, stream>>>(adjs, Wh1, Wh2, nw, attb);
    gemm_out_kernel <<<32, 256, 0, stream>>>(attb, WhbT, out);
}
